// NexusXLSTM_20461224198602
// MI455X (gfx1250) — compile-verified, matrix operand order checked by reading, not run on hardware
//
#include <hip/hip_runtime.h>
#include <cstdint>
#include <cstddef>

// ---------------------------------------------------------------------------
// Types / constants
// ---------------------------------------------------------------------------
typedef __attribute__((ext_vector_type(16))) __bf16 v16bf;
typedef __attribute__((ext_vector_type(8)))  float  v8f;

constexpr int BATCH   = 32;
constexpr int SEQ     = 2048;
constexpr int NFEAT   = 350;
constexpr int NFEAT_P = 352;   // padded to multiple of 32
constexpr int COMB_P  = 384;   // 350 + 32 regime emb, padded 382 -> 384
constexpr int DMODEL  = 512;
constexpr int DSTATE  = 64;
constexpr int NGATE   = 256;   // 4 * 64 gates concatenated
constexpr int MROWS   = BATCH * SEQ;  // 65536
constexpr float CLAMP_V = 20.0f;

// ---------------------------------------------------------------------------
// WMMA fragment loader.
// ISA 16-bit A-matrix 16x32 layout (CDNA5 ISA §7.12.2):
//   lane l<16  : row M=l,  K = {0..7} in VGPR0..3, {16..23} in VGPR4..7
//   lane l>=16 : row M=l-16, K = {8..15} and {24..31}
// With row-major source (stride ld), each lane needs two contiguous 16-byte
// chunks: [k0 + 8*hf, +8) and [k0 + 16 + 8*hf, +8)  -> two global_load_b128.
// B operand uses the same loader on the *transposed* weight (N x Kp).
// ---------------------------------------------------------------------------
__device__ __forceinline__ v16bf load_frag(const __bf16* __restrict__ p,
                                           int ld, int row0, int k0, int lane) {
    int r  = row0 + (lane & 15);
    int hf = (lane >> 4) & 1;
    const __bf16* base = p + (size_t)r * ld + k0 + 8 * hf;
    union { v16bf v; uint4 q[2]; } f;
    f.q[0] = *(const uint4*)(base);
    f.q[1] = *(const uint4*)(base + 16);
    return f.v;
}

// ---------------------------------------------------------------------------
// Generic WMMA GEMM:  C[M x N] = act( A[M x K]_bf16 * Wt[N x K]_bf16^T + bias )
// Block: 256 threads = 8 waves; block tile = 128 rows x 32 cols;
// each wave computes a 16x32 tile as two 16x16 v_wmma_f32_16x16x32_bf16 accums.
// epi: 0 = bias, 1 = bias+ELU, 2 = gate pack (clip/clip/tanh/sigmoid per 64)
// ---------------------------------------------------------------------------
__global__ __launch_bounds__(256) void wmma_gemm_kernel(
    const __bf16* __restrict__ A, int lda,
    const __bf16* __restrict__ Wt, int ldw,
    float* __restrict__ C, int ldc,
    const float* __restrict__ bias,
    int K, int epi) {
    const int lane = threadIdx.x & 31;
    const int wave = threadIdx.x >> 5;
    const int row0 = blockIdx.y * 128 + wave * 16;
    const int col0 = blockIdx.x * 32;

    v8f acc0 = {};
    v8f acc1 = {};
    for (int k0 = 0; k0 < K; k0 += 32) {
        v16bf a  = load_frag(A,  lda, row0,      k0, lane);
        v16bf b0 = load_frag(Wt, ldw, col0,      k0, lane);
        v16bf b1 = load_frag(Wt, ldw, col0 + 16, k0, lane);
        acc0 = __builtin_amdgcn_wmma_f32_16x16x32_bf16(
            false, a, false, b0, (short)0, acc0, false, false);
        acc1 = __builtin_amdgcn_wmma_f32_16x16x32_bf16(
            false, a, false, b1, (short)0, acc1, false, false);
    }

    // C/D layout: VGPR r -> M = r (lanes 0-15) or r+8 (lanes 16-31); N = lane&15
    const int hf = lane >> 4;
    const int m0 = row0 + hf * 8;
    const int nb = col0 + (lane & 15);
#pragma unroll
    for (int j = 0; j < 2; ++j) {
        const int n  = nb + j * 16;
        const float bv = bias[n];
#pragma unroll
        for (int r = 0; r < 8; ++r) {
            float v = (j ? acc1[r] : acc0[r]) + bv;
            if (epi == 1) {
                v = v > 0.0f ? v : expm1f(v);
            } else if (epi == 2) {
                const int g = n >> 6;
                if (g < 2)       v = fminf(CLAMP_V, fmaxf(-CLAMP_V, v));
                else if (g == 2) v = tanhf(v);
                else             v = 1.0f / (1.0f + __expf(-v));
            }
            C[(size_t)(m0 + r) * ldc + n] = v;
        }
    }
}

// ---------------------------------------------------------------------------
// Weight prep: transpose + pad + fp32->bf16.  Wt is Np x Kp, zero-padded.
// ---------------------------------------------------------------------------
__global__ void transpose_pad_bf16_kernel(const float* __restrict__ W,
                                          __bf16* __restrict__ Wt,
                                          int K, int N, int Kp, int Np) {
    int idx = blockIdx.x * blockDim.x + threadIdx.x;
    if (idx >= Np * Kp) return;
    int n = idx / Kp, k = idx - n * Kp;
    float v = (n < N && k < K) ? W[(size_t)k * N + n] : 0.0f;
    Wt[idx] = (__bf16)v;
}

// Pack the 4 gate projection weights into one transposed 256x512 bf16 matrix
// (cols [0,64)=i, [64,128)=f, [128,192)=z, [192,256)=o) + packed bias.
__global__ void gates_pack_kernel(const float* __restrict__ wi_,
                                  const float* __restrict__ wf_,
                                  const float* __restrict__ wz_,
                                  const float* __restrict__ wo_,
                                  const float* __restrict__ bi_,
                                  const float* __restrict__ bf_,
                                  const float* __restrict__ bz_,
                                  const float* __restrict__ bo_,
                                  __bf16* __restrict__ Wgt,
                                  float* __restrict__ gbias) {
    int idx = blockIdx.x * blockDim.x + threadIdx.x;
    if (idx >= NGATE * DMODEL) return;
    int n = idx >> 9;          // 0..255
    int k = idx & 511;         // 0..511
    int g = n >> 6, e = n & 63;
    const float* w = (g == 0) ? wi_ : (g == 1) ? wf_ : (g == 2) ? wz_ : wo_;
    Wgt[(size_t)n * DMODEL + k] = (__bf16)w[(size_t)k * DSTATE + e];
    if (k == 0) {
        const float* bb = (g == 0) ? bi_ : (g == 1) ? bf_ : (g == 2) ? bz_ : bo_;
        gbias[n] = bb[e];
    }
}

__global__ void pad_bias_kernel(const float* __restrict__ b, float* __restrict__ bp,
                                int n, int np) {
    int i = blockIdx.x * blockDim.x + threadIdx.x;
    if (i < np) bp[i] = (i < n) ? b[i] : 0.0f;
}

// ---------------------------------------------------------------------------
// comb = concat(x, emb_regime[regime_ids]) -> bf16, padded to 384 cols
// ---------------------------------------------------------------------------
__global__ void comb_kernel(const float* __restrict__ x,
                            const int* __restrict__ rid,
                            const float* __restrict__ emb,
                            __bf16* __restrict__ out) {
    size_t idx = (size_t)blockIdx.x * blockDim.x + threadIdx.x;
    if (idx >= (size_t)MROWS * COMB_P) return;
    int row = (int)(idx / COMB_P);
    int f   = (int)(idx - (size_t)row * COMB_P);
    float v;
    if (f < NFEAT)      v = x[(size_t)row * NFEAT + f];
    else if (f < 382)   v = emb[rid[row] * 32 + (f - NFEAT)];
    else                v = 0.0f;
    out[idx] = (__bf16)v;
}

__global__ void f32_to_bf16_kernel(const float* __restrict__ in,
                                   __bf16* __restrict__ out, size_t n) {
    size_t i = (size_t)blockIdx.x * blockDim.x + threadIdx.x;
    if (i < n) out[i] = (__bf16)in[i];
}

// ---------------------------------------------------------------------------
// LayerNorm (wave-per-row over 512) -> bf16 output
// ---------------------------------------------------------------------------
__global__ __launch_bounds__(256) void ln_bf16_kernel(
    const float* __restrict__ H,
    const float* __restrict__ g, const float* __restrict__ b,
    __bf16* __restrict__ out) {
    int row  = blockIdx.x * 8 + (threadIdx.x >> 5);
    int lane = threadIdx.x & 31;
    const float* x = H + (size_t)row * DMODEL;
    float v[16], s = 0.0f, s2 = 0.0f;
#pragma unroll
    for (int i = 0; i < 16; ++i) {
        float t = x[lane + 32 * i];
        v[i] = t; s += t; s2 += t * t;
    }
#pragma unroll
    for (int off = 16; off > 0; off >>= 1) {
        s  += __shfl_xor(s, off);
        s2 += __shfl_xor(s2, off);
    }
    float mu = s * (1.0f / DMODEL);
    float var = s2 * (1.0f / DMODEL) - mu * mu;
    float rs = rsqrtf(var + 1e-5f);
#pragma unroll
    for (int i = 0; i < 16; ++i) {
        int d = lane + 32 * i;
        out[(size_t)row * DMODEL + d] = (__bf16)((v[i] - mu) * rs * g[d] + b[d]);
    }
}

// h = LN(h + res) in place (fp32 master kept for residual precision)
__global__ __launch_bounds__(256) void ln_resid_kernel(
    float* __restrict__ H, const float* __restrict__ R,
    const float* __restrict__ g, const float* __restrict__ b) {
    int row  = blockIdx.x * 8 + (threadIdx.x >> 5);
    int lane = threadIdx.x & 31;
    float* x = H + (size_t)row * DMODEL;
    const float* r = R + (size_t)row * DMODEL;
    float v[16], s = 0.0f, s2 = 0.0f;
#pragma unroll
    for (int i = 0; i < 16; ++i) {
        int d = lane + 32 * i;
        float t = x[d] + r[d];
        v[i] = t; s += t; s2 += t * t;
    }
#pragma unroll
    for (int off = 16; off > 0; off >>= 1) {
        s  += __shfl_xor(s, off);
        s2 += __shfl_xor(s2, off);
    }
    float mu = s * (1.0f / DMODEL);
    float var = s2 * (1.0f / DMODEL) - mu * mu;
    float rs = rsqrtf(var + 1e-5f);
#pragma unroll
    for (int i = 0; i < 16; ++i) {
        int d = lane + 32 * i;
        x[d] = (v[i] - mu) * rs * g[d] + b[d];
    }
}

// ---------------------------------------------------------------------------
// softmax over 350 logits (wave-per-row) fused with xw = x * weights -> bf16,
// padded to 352; also emits weights[:, -1, :] to the output tuple slot.
// ---------------------------------------------------------------------------
__global__ __launch_bounds__(256) void softmax_xw_kernel(
    const float* __restrict__ logits,     // ld = 352
    const float* __restrict__ x,          // ld = 350
    __bf16* __restrict__ xw,              // ld = 352
    float* __restrict__ wlast) {          // [B, 350]
    int row  = blockIdx.x * 8 + (threadIdx.x >> 5);
    int lane = threadIdx.x & 31;
    const float* lg = logits + (size_t)row * NFEAT_P;
    float v[11], mx = -1e30f;
#pragma unroll
    for (int i = 0; i < 11; ++i) {
        int f = lane + 32 * i;
        float t = (f < NFEAT) ? lg[f] : -1e30f;
        v[i] = t; mx = fmaxf(mx, t);
    }
#pragma unroll
    for (int off = 16; off > 0; off >>= 1) mx = fmaxf(mx, __shfl_xor(mx, off));
    float s = 0.0f;
#pragma unroll
    for (int i = 0; i < 11; ++i) {
        int f = lane + 32 * i;
        float e = (f < NFEAT) ? __expf(v[i] - mx) : 0.0f;
        v[i] = e; s += e;
    }
#pragma unroll
    for (int off = 16; off > 0; off >>= 1) s += __shfl_xor(s, off);
    float inv = 1.0f / s;
    const bool last = ((row & (SEQ - 1)) == SEQ - 1);
    const int bidx = row >> 11;  // row / SEQ
#pragma unroll
    for (int i = 0; i < 11; ++i) {
        int f = lane + 32 * i;
        if (f < NFEAT_P) {
            float w = v[i] * inv;
            float xv = (f < NFEAT) ? x[(size_t)row * NFEAT + f] * w : 0.0f;
            xw[(size_t)row * NFEAT_P + f] = (__bf16)xv;
            if (last && f < NFEAT) wlast[(size_t)bidx * NFEAT + f] = w;
        }
    }
}

// ---------------------------------------------------------------------------
// sLSTM recurrent scan: 2048 lanes = (b, e); 2048 sequential steps each.
// gates (fully activated in the GEMM epilogue): [li | lf | z | o] at ld 256.
// ---------------------------------------------------------------------------
__global__ __launch_bounds__(256) void scan_kernel(
    const float* __restrict__ gates, __bf16* __restrict__ hs) {
    int t = blockIdx.x * blockDim.x + threadIdx.x;  // 0 .. 2047
    int b = t >> 6, e = t & 63;
    const float* gp = gates + (size_t)b * SEQ * NGATE + e;
    __bf16* hp = hs + (size_t)b * SEQ * DSTATE + e;
    float m = 0.0f, c = 0.0f;
    for (int s = 0; s < SEQ; ++s) {
        float li = gp[0], lf = gp[64], zt = gp[128], ot = gp[192];
        __builtin_prefetch(gp + 4 * NGATE, 0, 0);   // global_prefetch next steps
        float mn = fmaxf(m + lf, li);
        float i_s = __expf(li - mn);
        float f_s = __expf(m + lf - mn);
        c = f_s * c + i_s * zt;
        m = mn;
        *hp = (__bf16)(ot * tanhf(c));
        gp += NGATE;
        hp += DSTATE;
    }
}

// ---------------------------------------------------------------------------
// Output heads on h_last (B x 512): tiny, one block per batch row.
// d_out layout (floats): d15[32] | d30[32] | hv[96] | hreg[192] | hrng[96]
//                        | h_last[32*512] | w_last[32*350]
// ---------------------------------------------------------------------------
__global__ __launch_bounds__(64) void heads_kernel(
    const float* __restrict__ H,
    const float* __restrict__ w15,  const float* __restrict__ b15,
    const float* __restrict__ w30,  const float* __restrict__ b30,
    const float* __restrict__ wv,   const float* __restrict__ bv,
    const float* __restrict__ wreg, const float* __restrict__ breg,
    const float* __restrict__ wrng, const float* __restrict__ brng,
    float* __restrict__ out) {
    __shared__ float sh[DMODEL];
    const int b = blockIdx.x;
    const float* h = H + ((size_t)b * SEQ + (SEQ - 1)) * DMODEL;
    for (int i = threadIdx.x; i < DMODEL; i += blockDim.x) {
        float v = h[i];
        sh[i] = v;
        out[448 + b * DMODEL + i] = v;  // h_last
    }
    __syncthreads();
    const int t = threadIdx.x;
    if (t == 0) {
        float s = b15[0];
        for (int i = 0; i < DMODEL; ++i) s += sh[i] * w15[i];
        out[b] = s;
    } else if (t == 1) {
        float s = b30[0];
        for (int i = 0; i < DMODEL; ++i) s += sh[i] * w30[i];
        out[32 + b] = s;
    } else if (t >= 2 && t < 5) {
        int j = t - 2;
        float s = bv[j];
        for (int i = 0; i < DMODEL; ++i) s += sh[i] * wv[i * 3 + j];
        out[64 + b * 3 + j] = s;
    } else if (t >= 5 && t < 11) {
        int j = t - 5;
        float s = breg[j];
        for (int i = 0; i < DMODEL; ++i) s += sh[i] * wreg[i * 6 + j];
        out[160 + b * 6 + j] = s;
    } else if (t >= 11 && t < 14) {
        int j = t - 11;
        float s = brng[j];
        for (int i = 0; i < DMODEL; ++i) s += sh[i] * wrng[i * 3 + j];
        out[352 + b * 3 + j] = s;
    }
}

// ---------------------------------------------------------------------------
// Host-side orchestration
// ---------------------------------------------------------------------------
static inline dim3 grid1d(size_t n) { return dim3((unsigned)((n + 255) / 256)); }

extern "C" void kernel_launch(void* const* d_in, const int* in_sizes, int n_in,
                              void* d_out, int out_size, void* d_ws, size_t ws_size,
                              hipStream_t stream) {
    (void)in_sizes; (void)n_in; (void)out_size; (void)ws_size;
    // Inputs (setup_inputs order)
    const float* x        = (const float*)d_in[0];
    const int*   rids     = (const int*)  d_in[1];
    const float* emb      = (const float*)d_in[2];
    const float* grn_w1   = (const float*)d_in[3];
    const float* grn_b1   = (const float*)d_in[4];
    const float* grn_w2   = (const float*)d_in[5];
    const float* grn_b2   = (const float*)d_in[6];
    const float* in_w     = (const float*)d_in[7];
    const float* in_b     = (const float*)d_in[8];
    const float* ln_in_g  = (const float*)d_in[9];
    const float* ln_in_b  = (const float*)d_in[10];
    const float* wi_      = (const float*)d_in[11];
    const float* bi_      = (const float*)d_in[12];
    const float* wf_      = (const float*)d_in[13];
    const float* bfg      = (const float*)d_in[14];
    const float* wz_      = (const float*)d_in[15];
    const float* bz_      = (const float*)d_in[16];
    const float* wo_      = (const float*)d_in[17];
    const float* bo_      = (const float*)d_in[18];
    const float* wp_      = (const float*)d_in[19];
    const float* bp_      = (const float*)d_in[20];
    const float* nm_g     = (const float*)d_in[21];
    const float* nm_b     = (const float*)d_in[22];
    const float* hd15_w   = (const float*)d_in[23];
    const float* hd15_b   = (const float*)d_in[24];
    const float* hd30_w   = (const float*)d_in[25];
    const float* hd30_b   = (const float*)d_in[26];
    const float* hv_w     = (const float*)d_in[27];
    const float* hv_b     = (const float*)d_in[28];
    const float* hreg_w   = (const float*)d_in[29];
    const float* hreg_b   = (const float*)d_in[30];
    const float* hrng_w   = (const float*)d_in[31];
    const float* hrng_b   = (const float*)d_in[32];
    float* out = (float*)d_out;

    // Workspace carve-up (256B aligned)
    char* ws = (char*)d_ws;
    size_t off = 0;
    auto carve = [&](size_t bytes) {
        void* p = ws + off;
        off = (off + bytes + 255) & ~(size_t)255;
        return p;
    };
    __bf16* A_bf  = (__bf16*)carve((size_t)MROWS * DMODEL * 2);   // activations (bf16)
    float*  G     = (float*) carve((size_t)MROWS * DMODEL * 4);   // gemm out / gates / res
    float*  H     = (float*) carve((size_t)MROWS * DMODEL * 4);   // fp32 h master
    __bf16* HSbf  = (__bf16*)carve((size_t)MROWS * DSTATE * 2);   // scan outputs (bf16)
    __bf16* W1t   = (__bf16*)carve((size_t)DMODEL * COMB_P * 2);
    __bf16* W2t   = (__bf16*)carve((size_t)NFEAT_P * DMODEL * 2);
    __bf16* W3t   = (__bf16*)carve((size_t)DMODEL * NFEAT_P * 2);
    __bf16* Wgt   = (__bf16*)carve((size_t)NGATE * DMODEL * 2);
    __bf16* Wpt   = (__bf16*)carve((size_t)DMODEL * DSTATE * 2);
    float*  gbias = (float*) carve(NGATE * 4);
    float*  b2p   = (float*) carve(NFEAT_P * 4);

    float* out_wlast = out + 448 + BATCH * DMODEL;  // weights[:, -1, :]

    // --- weight prep ---
    transpose_pad_bf16_kernel<<<grid1d((size_t)DMODEL * COMB_P), 256, 0, stream>>>(
        grn_w1, W1t, 382, DMODEL, COMB_P, DMODEL);
    transpose_pad_bf16_kernel<<<grid1d((size_t)NFEAT_P * DMODEL), 256, 0, stream>>>(
        grn_w2, W2t, DMODEL, NFEAT, DMODEL, NFEAT_P);
    transpose_pad_bf16_kernel<<<grid1d((size_t)DMODEL * NFEAT_P), 256, 0, stream>>>(
        in_w, W3t, NFEAT, DMODEL, NFEAT_P, DMODEL);
    pad_bias_kernel<<<grid1d(NFEAT_P), 256, 0, stream>>>(grn_b2, b2p, NFEAT, NFEAT_P);

    // --- GRN: comb -> hidden (ELU) -> logits -> softmax*x -> input proj ---
    comb_kernel<<<grid1d((size_t)MROWS * COMB_P), 256, 0, stream>>>(x, rids, emb, A_bf);
    wmma_gemm_kernel<<<dim3(DMODEL / 32, MROWS / 128), 256, 0, stream>>>(
        A_bf, COMB_P, W1t, COMB_P, G, DMODEL, grn_b1, COMB_P, /*epi=*/1);
    f32_to_bf16_kernel<<<grid1d((size_t)MROWS * DMODEL), 256, 0, stream>>>(
        G, A_bf, (size_t)MROWS * DMODEL);
    wmma_gemm_kernel<<<dim3(NFEAT_P / 32, MROWS / 128), 256, 0, stream>>>(
        A_bf, DMODEL, W2t, DMODEL, G, NFEAT_P, b2p, DMODEL, /*epi=*/0);
    softmax_xw_kernel<<<dim3(MROWS / 8), 256, 0, stream>>>(G, x, A_bf, out_wlast);
    wmma_gemm_kernel<<<dim3(DMODEL / 32, MROWS / 128), 256, 0, stream>>>(
        A_bf, NFEAT_P, W3t, NFEAT_P, H, DMODEL, in_b, NFEAT_P, /*epi=*/0);

    // --- 4 sLSTM layers ---
    for (int l = 0; l < 4; ++l) {
        gates_pack_kernel<<<grid1d((size_t)NGATE * DMODEL), 256, 0, stream>>>(
            wi_ + (size_t)l * DMODEL * DSTATE, wf_ + (size_t)l * DMODEL * DSTATE,
            wz_ + (size_t)l * DMODEL * DSTATE, wo_ + (size_t)l * DMODEL * DSTATE,
            bi_ + l * DSTATE, bfg + l * DSTATE, bz_ + l * DSTATE, bo_ + l * DSTATE,
            Wgt, gbias);
        transpose_pad_bf16_kernel<<<grid1d((size_t)DMODEL * DSTATE), 256, 0, stream>>>(
            wp_ + (size_t)l * DSTATE * DMODEL, Wpt, DSTATE, DMODEL, DSTATE, DMODEL);

        ln_bf16_kernel<<<dim3(MROWS / 8), 256, 0, stream>>>(
            H, ln_in_g + l * DMODEL, ln_in_b + l * DMODEL, A_bf);
        wmma_gemm_kernel<<<dim3(NGATE / 32, MROWS / 128), 256, 0, stream>>>(
            A_bf, DMODEL, Wgt, DMODEL, G, NGATE, gbias, DMODEL, /*epi=*/2);
        scan_kernel<<<dim3((BATCH * DSTATE) / 256), 256, 0, stream>>>(G, HSbf);
        wmma_gemm_kernel<<<dim3(DMODEL / 32, MROWS / 128), 256, 0, stream>>>(
            HSbf, DSTATE, Wpt, DSTATE, G, DMODEL, bp_ + l * DMODEL, DSTATE, /*epi=*/0);
        ln_resid_kernel<<<dim3(MROWS / 8), 256, 0, stream>>>(
            H, G, nm_g + l * DMODEL, nm_b + l * DMODEL);
    }

    // --- output heads ---
    heads_kernel<<<dim3(BATCH), 64, 0, stream>>>(
        H, hd15_w, hd15_b, hd30_w, hd30_b, hv_w, hv_b,
        hreg_w, hreg_b, hrng_w, hrng_b, out);
}
